// NNPredictor_46016279609973
// MI455X (gfx1250) — matrix operand order repaired and checked by reading, not compiled, not run on hardware
//
#include <hip/hip_runtime.h>
#include <stdint.h>

typedef __attribute__((ext_vector_type(16))) __bf16 v16bf;
typedef __attribute__((ext_vector_type(8)))  float  v8f;
typedef __attribute__((ext_vector_type(2)))  float  vf2;
typedef __attribute__((ext_vector_type(2)))  __bf16 vb2;

#define N_EDGES 800000
#define N_NODES 50000
#define D_H 128
#define D_E 64
#define IN_FEATS 320   // 2*D_H + D_E
#define H1 256
#define H2 128
#define BN_EPS 1e-5f

// ---------------- helpers: native bf16 converts (v_cvt_pk_bf16_f32 path) ----------------
__device__ __forceinline__ uint16_t f2bf(float f) {
  union { uint16_t u; __bf16 h; } x;
  x.h = (__bf16)f;
  return x.u;
}
__device__ __forceinline__ float bf2f(uint16_t b) {
  union { uint16_t u; __bf16 h; } x;
  x.u = b;
  return (float)x.h;
}
__device__ __forceinline__ uint32_t pk_bf16(float a, float b) {
  vf2 f; f[0] = a; f[1] = b;
  union { vb2 v; uint32_t u; } x;
  x.v = __builtin_convertvector(f, vb2);   // -> single v_cvt_pk_bf16_f32
  return x.u;
}

// ---------------- stats zero-init ----------------
__global__ void zero_kernel(uint32_t* p, int n) {
  int i = blockIdx.x * blockDim.x + threadIdx.x;
  if (i < n) p[i] = 0u;
}

// ---------------- src/dst histograms ----------------
__global__ void hist_kernel(const int* __restrict__ src, const int* __restrict__ dst,
                            int* __restrict__ cnt_src, int* __restrict__ cnt_dst) {
  int i = blockIdx.x * blockDim.x + threadIdx.x;
  if (i < N_EDGES) {
    atomicAdd(&cnt_src[src[i]], 1);
    atomicAdd(&cnt_dst[dst[i]], 1);
  }
}

// column sums / sumsq of gathered h[src] (cols 0..127) and h[dst] (cols 128..255)
// via count-weighted pass over the (L2-resident) node table.
__global__ void __launch_bounds__(128) nodestats_kernel(const float* __restrict__ h,
                                                        const int* __restrict__ cs,
                                                        const int* __restrict__ cd,
                                                        float* __restrict__ sums0,
                                                        float* __restrict__ ssq0) {
  int c = threadIdx.x;
  float s_s = 0.f, q_s = 0.f, s_d = 0.f, q_d = 0.f;
  for (int n = blockIdx.x; n < N_NODES; n += gridDim.x) {
    float v = h[(size_t)n * D_H + c];
    float a = (float)cs[n], b = (float)cd[n];
    s_s += a * v;  q_s += a * v * v;
    s_d += b * v;  q_d += b * v * v;
  }
  atomicAdd(&sums0[c], s_s);        atomicAdd(&ssq0[c], q_s);
  atomicAdd(&sums0[D_H + c], s_d);  atomicAdd(&ssq0[D_H + c], q_d);
}

// column sums / sumsq of e (cols 256..319 of x0)
__global__ void __launch_bounds__(256) estats_kernel(const float* __restrict__ e,
                                                     float* __restrict__ sums0,
                                                     float* __restrict__ ssq0) {
  int c = threadIdx.x & 63;
  int rl = threadIdx.x >> 6;
  float s = 0.f, q = 0.f;
  for (int r = blockIdx.x * 4 + rl; r < N_EDGES; r += gridDim.x * 4) {
    float v = e[(size_t)r * D_E + c];
    s += v; q += v * v;
  }
  atomicAdd(&sums0[2 * D_H + c], s);
  atomicAdd(&ssq0[2 * D_H + c], q);
}

// column stats of a bf16 activation matrix; blockDim.x == ncols
__global__ void colstats_kernel(const uint16_t* __restrict__ x,
                                float* __restrict__ sums, float* __restrict__ ssq,
                                int ncols) {
  int c = threadIdx.x;
  float s = 0.f, q = 0.f;
  for (int r = blockIdx.x; r < N_EDGES; r += gridDim.x) {
    float v = bf2f(x[(size_t)r * ncols + c]);
    s += v; q += v * v;
  }
  atomicAdd(&sums[c], s);
  atomicAdd(&ssq[c], q);
}

// BN batch-stats -> per-column affine: scale = g*rsqrt(var+eps), shift = b - mean*scale
__global__ void bnparams_kernel(const float* __restrict__ sums, const float* __restrict__ ssq,
                                const float* __restrict__ g, const float* __restrict__ b,
                                float* __restrict__ scale, float* __restrict__ shift, int n) {
  int j = blockIdx.x * blockDim.x + threadIdx.x;
  if (j < n) {
    const float invE = 1.0f / (float)N_EDGES;
    float mean = sums[j] * invE;
    float var  = ssq[j] * invE - mean * mean;
    float sc   = g[j] * rsqrtf(var + BN_EPS);
    scale[j] = sc;
    shift[j] = b[j] - mean * sc;
  }
}

// fold BN into next layer:  Wf = bf16(W * diag(scale)),  bfold = b + W @ shift
__global__ void __launch_bounds__(128) fold_kernel(const float* __restrict__ W,
                                                   const float* __restrict__ b,
                                                   const float* __restrict__ scale,
                                                   const float* __restrict__ shift,
                                                   uint16_t* __restrict__ Wf,
                                                   float* __restrict__ bfold, int in_dim) {
  __shared__ float red[128];
  int o = blockIdx.x;
  float acc = 0.f;
  for (int j = threadIdx.x; j < in_dim; j += 128) {
    float w = W[(size_t)o * in_dim + j];
    acc += w * shift[j];
    Wf[(size_t)o * in_dim + j] = f2bf(w * scale[j]);
  }
  red[threadIdx.x] = acc;
  __syncthreads();
  for (int s = 64; s > 0; s >>= 1) {
    if (threadIdx.x < s) red[threadIdx.x] += red[threadIdx.x + s];
    __syncthreads();
  }
  if (threadIdx.x == 0) bfold[o] = b[o] + red[0];
}

// ---------------- GEMM1: y1 = ReLU(gather(x0) @ W1f.T + b1f), bf16 out ----------------
// block = 256 threads (8 waves), tile = 64 rows x 64 cols, K = 320 in steps of 32.
// LDS ping-pong: one barrier per k-step; next tile's gather loads overlap compute.
__global__ void __launch_bounds__(256) gemm1_kernel(const float* __restrict__ h,
                                                    const float* __restrict__ e,
                                                    const int* __restrict__ src,
                                                    const int* __restrict__ dst,
                                                    const uint16_t* __restrict__ W1f,
                                                    const float* __restrict__ b1f,
                                                    uint16_t* __restrict__ y1) {
  __shared__ uint16_t As[2][64 * 34];
  const int t = threadIdx.x;
  const int row_base = blockIdx.x * 64;
  const int col_base = blockIdx.y * 64;
  const int lane = t & 31;
  const int w = t >> 5;
  const int half = lane >> 4;
  const int mn = lane & 15;
  const int m_tile = (w & 3) * 16;
  const int n_tile = col_base + (w >> 2) * 32;

  const int r = t >> 2, seg = t & 3;
  const int gr = row_base + r;
  const int si = src[gr], di = dst[gr];

  const float* srcA = h + (size_t)si * D_H;      // kb in [0,128)
  const float* srcB = h + (size_t)di * D_H;      // kb in [128,256)
  const float* srcE = e + (size_t)gr * D_E;      // kb in [256,320)

  v8f acc0 = {}; v8f acc1 = {};

  // preload k-tile 0
  float4 f0 = ((const float4*)(srcA + seg * 8))[0];
  float4 f1 = ((const float4*)(srcA + seg * 8))[1];

#pragma unroll
  for (int i = 0; i < IN_FEATS / 32; ++i) {
    const int kb = i * 32;
    const int cur = i & 1;
    // ---- stage current tile (registers -> LDS, hw packed converts) ----
    uint32_t p0 = pk_bf16(f0.x, f0.y), p1 = pk_bf16(f0.z, f0.w);
    uint32_t p2 = pk_bf16(f1.x, f1.y), p3 = pk_bf16(f1.z, f1.w);
    uint32_t* dsp = (uint32_t*)&As[cur][r * 34 + seg * 8];
    dsp[0] = p0; dsp[1] = p1; dsp[2] = p2; dsp[3] = p3;
    // ---- issue next tile's gather loads (overlap with compute) ----
    if (i + 1 < IN_FEATS / 32) {
      const int kn = kb + 32;
      const float* p;
      if (kn < D_H)            p = srcA + kn;
      else if (kn < 2 * D_H)   p = srcB + (kn - D_H);
      else                     p = srcE + (kn - 2 * D_H);
      f0 = ((const float4*)(p + seg * 8))[0];
      f1 = ((const float4*)(p + seg * 8))[1];
    }
    __syncthreads();  // buf[cur] visible; closes reads of buf[!cur] from step i-1

    // ---- fragments + WMMA ----
    union { uint32_t u[8]; v16bf v; } fa, fb0, fb1;
#pragma unroll
    for (int vv = 0; vv < 8; ++vv) {
      int ka = 16 * half + 2 * vv;
      fa.u[vv] = *(const uint32_t*)&As[cur][(m_tile + mn) * 34 + ka];
      int kbb = 16 * half + 2 * vv;
      fb0.u[vv] = *(const uint32_t*)&W1f[(size_t)(n_tile + mn) * IN_FEATS + kb + kbb];
      fb1.u[vv] = *(const uint32_t*)&W1f[(size_t)(n_tile + 16 + mn) * IN_FEATS + kb + kbb];
    }
    acc0 = __builtin_amdgcn_wmma_f32_16x16x32_bf16(false, fa.v, false, fb0.v, (short)0, acc0, false, false);
    acc1 = __builtin_amdgcn_wmma_f32_16x16x32_bf16(false, fa.v, false, fb1.v, (short)0, acc1, false, false);
  }

  // ---- epilogue: bias + ReLU, store bf16 ----
  float bias0 = b1f[n_tile + mn];
  float bias1 = b1f[n_tile + 16 + mn];
#pragma unroll
  for (int rr = 0; rr < 8; ++rr) {
    int rowm = row_base + m_tile + rr + 8 * half;
    float v0 = acc0[rr] + bias0; v0 = v0 > 0.f ? v0 : 0.f;
    float v1 = acc1[rr] + bias1; v1 = v1 > 0.f ? v1 : 0.f;
    y1[(size_t)rowm * H1 + n_tile + mn]      = f2bf(v0);
    y1[(size_t)rowm * H1 + n_tile + 16 + mn] = f2bf(v1);
  }
}

// ---------------- GEMM2: y2 = ReLU(y1 @ W2f.T + b2f), bf16 in/out ----------------
// A tiles staged with CDNA5 async global->LDS DMA (ASYNCcnt), double-buffered so the
// copy of tile i+1 overlaps the WMMA work on tile i.
#define LDSW 40  // row stride in halves: 80B rows keep every 16B async B128 dest aligned
__global__ void __launch_bounds__(256) gemm2_kernel(const uint16_t* __restrict__ y1,
                                                    const uint16_t* __restrict__ W2f,
                                                    const float* __restrict__ b2f,
                                                    uint16_t* __restrict__ y2) {
  __shared__ uint16_t As[2][64 * LDSW];
  const int t = threadIdx.x;
  const int row_base = blockIdx.x * 64;
  const int col_base = blockIdx.y * 64;
  const int lane = t & 31;
  const int w = t >> 5;
  const int half = lane >> 4;
  const int mn = lane & 15;
  const int m_tile = (w & 3) * 16;
  const int n_tile = col_base + (w >> 2) * 32;

  const int r = t >> 2, seg = t & 3;
  const int gr = row_base + r;

  const uint32_t lds_dst0 = (uint32_t)(uintptr_t)&As[0][r * LDSW + seg * 8];  // 16B aligned
  const uint32_t lds_dst1 = (uint32_t)(uintptr_t)&As[1][r * LDSW + seg * 8];
  const uint16_t* grow = y1 + (size_t)gr * H1 + seg * 8;

  v8f acc0 = {}; v8f acc1 = {};

  // kick off tile 0 copy
  {
    const uint64_t gsrc = (uint64_t)(uintptr_t)grow;
    asm volatile("global_load_async_to_lds_b128 %0, %1, off"
                 :: "v"(lds_dst0), "v"(gsrc) : "memory");
  }

#pragma unroll
  for (int i = 0; i < H1 / 32; ++i) {
    const int kb = i * 32;
    const int cur = i & 1;
    asm volatile("s_wait_asynccnt 0" ::: "memory");  // tile i's DMA complete
    __syncthreads();  // tile i visible to all waves; step i-1 reads of buf[!cur] done
    if (i + 1 < H1 / 32) {
      const uint64_t gsrc = (uint64_t)(uintptr_t)(grow + kb + 32);
      asm volatile("global_load_async_to_lds_b128 %0, %1, off"
                   :: "v"(cur ? lds_dst0 : lds_dst1), "v"(gsrc) : "memory");
    }

    union { uint32_t u[8]; v16bf v; } fa, fb0, fb1;
#pragma unroll
    for (int vv = 0; vv < 8; ++vv) {
      int ka = 16 * half + 2 * vv;
      fa.u[vv] = *(const uint32_t*)&As[cur][(m_tile + mn) * LDSW + ka];
      int kbb = 16 * half + 2 * vv;
      fb0.u[vv] = *(const uint32_t*)&W2f[(size_t)(n_tile + mn) * H1 + kb + kbb];
      fb1.u[vv] = *(const uint32_t*)&W2f[(size_t)(n_tile + 16 + mn) * H1 + kb + kbb];
    }
    acc0 = __builtin_amdgcn_wmma_f32_16x16x32_bf16(false, fa.v, false, fb0.v, (short)0, acc0, false, false);
    acc1 = __builtin_amdgcn_wmma_f32_16x16x32_bf16(false, fa.v, false, fb1.v, (short)0, acc1, false, false);
  }

  float bias0 = b2f[n_tile + mn];
  float bias1 = b2f[n_tile + 16 + mn];
#pragma unroll
  for (int rr = 0; rr < 8; ++rr) {
    int rowm = row_base + m_tile + rr + 8 * half;
    float v0 = acc0[rr] + bias0; v0 = v0 > 0.f ? v0 : 0.f;
    float v1 = acc1[rr] + bias1; v1 = v1 > 0.f ? v1 : 0.f;
    y2[(size_t)rowm * H2 + n_tile + mn]      = f2bf(v0);
    y2[(size_t)rowm * H2 + n_tile + 16 + mn] = f2bf(v1);
  }
}

// ---------------- final: out = BN2(y2) @ W3.T + b3  (folded: y2 . w3f + b3f) ----------------
__global__ void __launch_bounds__(256) final_kernel(const uint16_t* __restrict__ y2,
                                                    const uint16_t* __restrict__ w3f,
                                                    const float* __restrict__ b3f,
                                                    float* __restrict__ out) {
  int wv = threadIdx.x >> 5, lane = threadIdx.x & 31;
  size_t edge = (size_t)blockIdx.x * 8 + wv;
  const uint16_t* p = y2 + edge * H2 + lane * 4;
  float s = 0.f;
#pragma unroll
  for (int i = 0; i < 4; ++i) s += bf2f(p[i]) * bf2f(w3f[lane * 4 + i]);
  for (int off = 16; off > 0; off >>= 1) s += __shfl_down(s, off, 32);
  if (lane == 0) out[edge] = s + b3f[0];
}

// ---------------- launch ----------------
extern "C" void kernel_launch(void* const* d_in, const int* in_sizes, int n_in,
                              void* d_out, int out_size, void* d_ws, size_t ws_size,
                              hipStream_t stream) {
  const float* h     = (const float*)d_in[0];
  const float* e     = (const float*)d_in[1];
  const int*   src   = (const int*)d_in[2];
  const int*   dst   = (const int*)d_in[3];
  const float* bn0_g = (const float*)d_in[4];
  const float* bn0_b = (const float*)d_in[5];
  const float* W1    = (const float*)d_in[6];
  const float* b1    = (const float*)d_in[7];
  const float* bn1_g = (const float*)d_in[8];
  const float* bn1_b = (const float*)d_in[9];
  const float* W2    = (const float*)d_in[10];
  const float* b2    = (const float*)d_in[11];
  const float* bn2_g = (const float*)d_in[12];
  const float* bn2_b = (const float*)d_in[13];
  const float* W3    = (const float*)d_in[14];
  const float* b3    = (const float*)d_in[15];
  float* out = (float*)d_out;
  char* ws = (char*)d_ws;

  // workspace layout
  constexpr size_t OFF_Y1 = 0;
  constexpr size_t OFF_Y2 = OFF_Y1 + (size_t)N_EDGES * H1 * 2;
  constexpr size_t OFF_Z  = OFF_Y2 + (size_t)N_EDGES * H2 * 2;  // zeroed zone start
  constexpr size_t OFF_CS = OFF_Z;
  constexpr size_t OFF_CD = OFF_CS + (size_t)N_NODES * 4;
  constexpr size_t OFF_S0 = OFF_CD + (size_t)N_NODES * 4;
  constexpr size_t OFF_Q0 = OFF_S0 + IN_FEATS * 4;
  constexpr size_t OFF_S1 = OFF_Q0 + IN_FEATS * 4;
  constexpr size_t OFF_Q1 = OFF_S1 + H1 * 4;
  constexpr size_t OFF_S2 = OFF_Q1 + H1 * 4;
  constexpr size_t OFF_Q2 = OFF_S2 + H2 * 4;
  constexpr size_t ZERO_END = OFF_Q2 + H2 * 4;
  constexpr int ZERO_WORDS = (int)((ZERO_END - OFF_Z) / 4);
  constexpr size_t OFF_P   = (ZERO_END + 255) & ~(size_t)255;
  constexpr size_t OFF_SC0 = OFF_P;
  constexpr size_t OFF_SH0 = OFF_SC0 + IN_FEATS * 4;
  constexpr size_t OFF_SC1 = OFF_SH0 + IN_FEATS * 4;
  constexpr size_t OFF_SH1 = OFF_SC1 + H1 * 4;
  constexpr size_t OFF_SC2 = OFF_SH1 + H1 * 4;
  constexpr size_t OFF_SH2 = OFF_SC2 + H2 * 4;
  constexpr size_t OFF_B1F = OFF_SH2 + H2 * 4;
  constexpr size_t OFF_B2F = OFF_B1F + H1 * 4;
  constexpr size_t OFF_B3F = OFF_B2F + H2 * 4;
  constexpr size_t OFF_W1F = (OFF_B3F + 4 + 255) & ~(size_t)255;
  constexpr size_t OFF_W2F = OFF_W1F + (size_t)H1 * IN_FEATS * 2;
  constexpr size_t OFF_W3F = OFF_W2F + (size_t)H2 * H1 * 2;

  uint16_t* y1 = (uint16_t*)(ws + OFF_Y1);
  uint16_t* y2 = (uint16_t*)(ws + OFF_Y2);
  int* cnt_src = (int*)(ws + OFF_CS);
  int* cnt_dst = (int*)(ws + OFF_CD);
  float* s0 = (float*)(ws + OFF_S0); float* q0 = (float*)(ws + OFF_Q0);
  float* s1 = (float*)(ws + OFF_S1); float* q1 = (float*)(ws + OFF_Q1);
  float* s2 = (float*)(ws + OFF_S2); float* q2 = (float*)(ws + OFF_Q2);
  float* sc0 = (float*)(ws + OFF_SC0); float* sh0 = (float*)(ws + OFF_SH0);
  float* sc1 = (float*)(ws + OFF_SC1); float* sh1 = (float*)(ws + OFF_SH1);
  float* sc2 = (float*)(ws + OFF_SC2); float* sh2 = (float*)(ws + OFF_SH2);
  float* b1f = (float*)(ws + OFF_B1F);
  float* b2f = (float*)(ws + OFF_B2F);
  float* b3f = (float*)(ws + OFF_B3F);
  uint16_t* W1f = (uint16_t*)(ws + OFF_W1F);
  uint16_t* W2f = (uint16_t*)(ws + OFF_W2F);
  uint16_t* w3f = (uint16_t*)(ws + OFF_W3F);

  // 1) zero accumulators
  zero_kernel<<<(ZERO_WORDS + 255) / 256, 256, 0, stream>>>((uint32_t*)(ws + OFF_Z), ZERO_WORDS);
  // 2) BN0 stats via histograms + weighted node pass + e pass
  hist_kernel<<<(N_EDGES + 255) / 256, 256, 0, stream>>>(src, dst, cnt_src, cnt_dst);
  nodestats_kernel<<<512, 128, 0, stream>>>(h, cnt_src, cnt_dst, s0, q0);
  estats_kernel<<<1024, 256, 0, stream>>>(e, s0, q0);
  bnparams_kernel<<<(IN_FEATS + 127) / 128, 128, 0, stream>>>(s0, q0, bn0_g, bn0_b, sc0, sh0, IN_FEATS);
  // 3) fold BN0 into W1, run GEMM1
  fold_kernel<<<H1, 128, 0, stream>>>(W1, b1, sc0, sh0, W1f, b1f, IN_FEATS);
  gemm1_kernel<<<dim3(N_EDGES / 64, H1 / 64), 256, 0, stream>>>(h, e, src, dst, W1f, b1f, y1);
  // 4) BN1 stats, fold into W2, GEMM2
  colstats_kernel<<<2048, H1, 0, stream>>>(y1, s1, q1, H1);
  bnparams_kernel<<<(H1 + 127) / 128, 128, 0, stream>>>(s1, q1, bn1_g, bn1_b, sc1, sh1, H1);
  fold_kernel<<<H2, 128, 0, stream>>>(W2, b2, sc1, sh1, W2f, b2f, H1);
  gemm2_kernel<<<dim3(N_EDGES / 64, H2 / 64), 256, 0, stream>>>(y1, W2f, b2f, y2);
  // 5) BN2 stats, fold into W3, final dot
  colstats_kernel<<<2048, H2, 0, stream>>>(y2, s2, q2, H2);
  bnparams_kernel<<<1, 128, 0, stream>>>(s2, q2, bn2_g, bn2_b, sc2, sh2, H2);
  fold_kernel<<<1, 128, 0, stream>>>(W3, b3, sc2, sh2, w3f, b3f, H2);
  final_kernel<<<N_EDGES / 8, 256, 0, stream>>>(y2, w3f, b3f, out);

  (void)in_sizes; (void)n_in; (void)out_size; (void)ws_size;
}
